// MultiHeadAttention_46342697124168
// MI455X (gfx1250) — matrix operand order repaired and checked by reading, not compiled, not run on hardware
//
#include <hip/hip_runtime.h>
#include <hip/hip_bf16.h>

// ---------------------------------------------------------------------------
// Multi-head cross-attention for gfx1250 (CDNA5, wave32, WMMA + TDM).
// bf16 operands, f32 accumulation via v_wmma_f32_16x16x32_bf16.
// Shared tiles (GEMM weights, attention K/V) staged to LDS by the Tensor
// Data Mover, double-buffered and overlapped with WMMA compute.
// ---------------------------------------------------------------------------

typedef __attribute__((ext_vector_type(16))) __bf16 bf16x16;
typedef __attribute__((ext_vector_type(8)))  __bf16 bf16x8;
typedef __attribute__((ext_vector_type(4)))  __bf16 bf16x4;
typedef __attribute__((ext_vector_type(8)))  float  f32x8;
typedef __attribute__((ext_vector_type(4)))  unsigned int u32x4;
typedef __attribute__((ext_vector_type(4)))  int i32x4;
typedef __attribute__((ext_vector_type(8)))  int i32x8;

#define D_MODEL 1024
#define NHEAD   16
#define DHEAD   64
#define SEQ     2048
#define BATCH   2
#define MROWS   (BATCH * SEQ)   // 4096

#if __has_builtin(__builtin_amdgcn_tensor_load_to_lds)
#define HAVE_TDM 1
#else
#define HAVE_TDM 0
#endif

// Build a 16-element bf16 A-operand from two contiguous 8-element chunks.
// ISA 16-bit A layout (16x32 MxK): lane holds row m=lane%16, half h=lane/16;
// vector elem e<8 -> K = 8h+e ; e>=8 -> K = 16+8h+(e-8). Two 16B loads/lane.
__device__ __forceinline__ bf16x16 load_a_pair(const __bf16* lo, const __bf16* hi) {
    bf16x8 l = *(const bf16x8*)lo;
    bf16x8 h = *(const bf16x8*)hi;
    return __builtin_shufflevector(l, h, 0, 1, 2, 3, 4, 5, 6, 7,
                                         8, 9, 10, 11, 12, 13, 14, 15);
}

__device__ __forceinline__ f32x8 wmma_bf16(bf16x16 a, bf16x16 b, f32x8 c) {
    return __builtin_amdgcn_wmma_f32_16x16x32_bf16(
        /*neg_a=*/false, a, /*neg_b=*/false, b,
        /*c_mod=*/(short)0, c, /*reuse_a=*/false, /*reuse_b=*/false);
}

#if HAVE_TDM
// Issue a TDM 2D tile load (tile_x x tile_y elements of 2 bytes) from a
// row-major tensor (row stride = stride_x elements) into LDS at lds_byte.
// D# packing per CDNA5 ISA. Tracked by TENSORcnt.
__device__ __forceinline__ void tdm_load_tile_2d(
    unsigned lds_byte, const void* gptr,
    unsigned tile_x, unsigned tile_y,
    unsigned tensor_x, unsigned tensor_y, unsigned stride_x) {
    unsigned long long ga = (unsigned long long)gptr;
    u32x4 g0;
    g0[0] = 1u;                                    // count=1 (valid), user mode
    g0[1] = lds_byte;                              // lds_addr
    g0[2] = (unsigned)(ga & 0xffffffffu);          // global_addr[31:0]
    g0[3] = (unsigned)((ga >> 32) & 0x01ffffffu)   // global_addr[56:32]
            | (2u << 30);                          // type = 2 ("image")
    i32x8 g1;
    g1[0] = (int)(1u << 16);                       // data_size=1 -> 2 bytes
    g1[1] = (int)((tensor_x & 0xffffu) << 16);     // tensor_dim0[15:0] @ b63:48
    g1[2] = (int)(((tensor_x >> 16) & 0xffffu) |
                  ((tensor_y & 0xffffu) << 16));   // dim0 hi | dim1 lo
    g1[3] = (int)(((tensor_y >> 16) & 0xffffu) |
                  (tile_x << 16));                 // dim1 hi | tile_dim0
    g1[4] = (int)(tile_y & 0xffffu);               // tile_dim1 | tile_dim2=0
    g1[5] = (int)stride_x;                         // tensor_dim0_stride[31:0]
    g1[6] = 0;                                     // stride0 hi | stride1 lo
    g1[7] = 0;                                     // stride1 hi
    i32x4 z4 = {0, 0, 0, 0};
#if defined(__clang_major__) && __clang_major__ >= 23
    i32x8 z8 = {0, 0, 0, 0, 0, 0, 0, 0};
    __builtin_amdgcn_tensor_load_to_lds(g0, g1, z4, z4, z8, 0);
#else
    __builtin_amdgcn_tensor_load_to_lds(g0, g1, z4, z4, 0);
#endif
}
#endif

__device__ __forceinline__ unsigned lds_addr_of(const void* p) {
    return (unsigned)(unsigned long long)p;   // low 32 bits = LDS byte offset
}

// ---------------------------------------------------------------------------
// fp32 -> bf16 conversion (4 elements / thread).
// ---------------------------------------------------------------------------
__global__ __launch_bounds__(256) void cvt_f32_bf16(
    const float* __restrict__ src, __bf16* __restrict__ dst, int n4) {
    int i = blockIdx.x * blockDim.x + threadIdx.x;
    if (i < n4) {
        float4 v = ((const float4*)src)[i];
        bf16x4 o;
        o[0] = (__bf16)v.x; o[1] = (__bf16)v.y;
        o[2] = (__bf16)v.z; o[3] = (__bf16)v.w;
        ((bf16x4*)dst)[i] = o;
    }
}

// ---------------------------------------------------------------------------
// GEMM: C = A[4096,1024] @ W[1024,1024] (+bias), WMMA-tiled.
// Block = 256 threads = 8 waves; block tile 128x64. The shared 32x64 weight
// tile is staged in LDS (double-buffered): wave 0 issues a TDM DMA for the
// next k-step while all 8 waves compute from the current LDS tile.
// Epilogue modes: 0=Q (scale 0.125, head-split), 1=K (head-split transposed),
// 2=V (head-split), 3=final fp32 output.
// ---------------------------------------------------------------------------
__global__ __launch_bounds__(256) void gemm_bf16(
    const __bf16* __restrict__ A,
    const __bf16* __restrict__ W,
    const float*  __restrict__ bias,
    void* __restrict__ out, int mode) {
    __shared__ __bf16 btile[2][32 * 64];   // 2 x 4 KB double buffer

    const int lane = threadIdx.x & 31;
    const int wave = threadIdx.x >> 5;
    const int row0 = blockIdx.y * 128 + wave * 16;
    const int col0 = blockIdx.x * 64;
    const int am = lane & 15;   // A-layout row
    const int ah = lane >> 4;   // A-layout half

    const __bf16* arow = A + (size_t)(row0 + am) * D_MODEL;

    f32x8 acc0 = {}, acc1 = {}, acc2 = {}, acc3 = {};

#if HAVE_TDM
    if (wave == 0) {
        tdm_load_tile_2d(lds_addr_of(&btile[0][0]),
                         W + col0, 64, 32, D_MODEL, D_MODEL, D_MODEL);
    }
#endif
    for (int it = 0; it < D_MODEL / 32; ++it) {
        const int k0 = it * 32;
        const int cur = it & 1;
#if HAVE_TDM
        if (wave == 0) __builtin_amdgcn_s_wait_tensorcnt((short)0);
        __syncthreads();
        if (wave == 0 && it + 1 < D_MODEL / 32) {
            tdm_load_tile_2d(lds_addr_of(&btile[1 - cur][0]),
                             W + (size_t)(k0 + 32) * D_MODEL + col0,
                             64, 32, D_MODEL, D_MODEL, D_MODEL);
        }
#else
        // Fallback: cooperative staging, 256 lanes x 16B = 4 KB tile.
        {
            const int t = threadIdx.x;           // 0..255
            const int rr = t >> 3, cc = (t & 7) * 8;
            *(bf16x8*)&btile[cur][rr * 64 + cc] =
                *(const bf16x8*)(W + (size_t)(k0 + rr) * D_MODEL + col0 + cc);
            __syncthreads();
        }
#endif
        bf16x16 a = load_a_pair(arow + k0 + 8 * ah, arow + k0 + 16 + 8 * ah);
        if (k0 + 32 < D_MODEL)
            __builtin_prefetch((const void*)(arow + k0 + 32), 0, 3);
        // B layout (32x16 KxN): lane holds column n = lane%16; elem s -> K in table order.
        const __bf16* brow = &btile[cur][lane & 15];
        bf16x16 b0, b1, b2, b3;
#pragma unroll
        for (int s = 0; s < 16; ++s) {
            const int kk = (s < 8) ? (8 * ah + s) : (16 + 8 * ah + s - 8);
            b0[s] = brow[kk * 64];
            b1[s] = brow[kk * 64 + 16];
            b2[s] = brow[kk * 64 + 32];
            b3[s] = brow[kk * 64 + 48];
        }
        acc0 = wmma_bf16(a, b0, acc0);
        acc1 = wmma_bf16(a, b1, acc1);
        acc2 = wmma_bf16(a, b2, acc2);
        acc3 = wmma_bf16(a, b3, acc3);
        __syncthreads();   // everyone done with btile[cur] before it+2 overwrite
    }

    // C/D layout: element r of lane -> row m = r + 8*(lane/16), col n = lane%16.
    f32x8 accs[4] = {acc0, acc1, acc2, acc3};
#pragma unroll
    for (int j = 0; j < 4; ++j) {
        const int n  = col0 + 16 * j + (lane & 15);
        const float bn = bias[n];
#pragma unroll
        for (int r = 0; r < 8; ++r) {
            const int m = row0 + r + 8 * (lane >> 4);
            float val = accs[j][r] + bn;
            const int b = m >> 11;        // batch
            const int s = m & (SEQ - 1);  // position within sequence
            const int h = n >> 6;         // head
            const int d = n & (DHEAD - 1);
            if (mode == 0) {        // Q: [B,H,SQ,64], fold in 1/sqrt(64)
                ((__bf16*)out)[((size_t)((b * NHEAD + h) * SEQ + s) << 6) + d] =
                    (__bf16)(val * 0.125f);
            } else if (mode == 1) { // K transposed: [B,H,64,SK]
                ((__bf16*)out)[((size_t)((b * NHEAD + h) * DHEAD + d) * SEQ) + s] =
                    (__bf16)val;
            } else if (mode == 2) { // V: [B,H,SK,64]
                ((__bf16*)out)[((size_t)((b * NHEAD + h) * SEQ + s) << 6) + d] =
                    (__bf16)val;
            } else {                // final fp32 output [4096,1024]
                ((float*)out)[(size_t)m * D_MODEL + n] = val;
            }
        }
    }
}

// ---------------------------------------------------------------------------
// Flash attention. Grid = (SEQ/128, B*H), 256 threads = 8 waves.
// Each wave owns 16 query rows x full 64-d head; the block streams keys 32 at
// a time. The shared Kt (64x32) and V (32x64) tiles are TDM-staged into LDS
// (double-buffered) once per block instead of 8x redundant global loads.
// Per 32-key step per wave: 4 score WMMAs + online softmax + LDS relayout of
// P (C/D layout -> A layout, intra-wave) + 4 P@V WMMAs.
// ---------------------------------------------------------------------------
__global__ __launch_bounds__(256) void flash_attn(
    const __bf16* __restrict__ q16,   // [B*H, SEQ, 64] pre-scaled
    const __bf16* __restrict__ kt16,  // [B*H, 64, SEQ]
    const __bf16* __restrict__ v16,   // [B*H, SEQ, 64]
    __bf16* __restrict__ ctx16) {     // [4096, 1024]
    __shared__ __bf16 ktile[2][64 * 32];  // 2 x 4 KB: [d=64][key=32]
    __shared__ __bf16 vtile[2][32 * 64];  // 2 x 4 KB: [key=32][d=64]
    __shared__ __bf16 plds[8][16 * 32];   // 1 KB per wave (P relayout)

    const int lane = threadIdx.x & 31;
    const int wave = threadIdx.x >> 5;
    const int bh = blockIdx.y;
    const int q0 = blockIdx.x * 128 + wave * 16;
    const int am = lane & 15;
    const int ah = lane >> 4;

    const __bf16* Q  = q16  + ((size_t)bh * SEQ + q0) * DHEAD;
    const __bf16* Kt = kt16 + (size_t)bh * DHEAD * SEQ;
    const __bf16* V  = v16  + (size_t)bh * SEQ * DHEAD;
    __bf16* lw = plds[wave];

    // Q tile is loop-invariant: A operands for d=[0,32) and d=[32,64).
    bf16x16 a0 = load_a_pair(Q + am * DHEAD + 8 * ah,      Q + am * DHEAD + 16 + 8 * ah);
    bf16x16 a1 = load_a_pair(Q + am * DHEAD + 32 + 8 * ah, Q + am * DHEAD + 48 + 8 * ah);

    float mrow[8], lsum[8];
#pragma unroll
    for (int r = 0; r < 8; ++r) { mrow[r] = -3.0e38f; lsum[r] = 0.0f; }
    f32x8 o0 = {}, o1 = {}, o2 = {}, o3 = {};

#if HAVE_TDM
    if (wave == 0) {
        tdm_load_tile_2d(lds_addr_of(&ktile[0][0]), Kt, 32, 64, SEQ, DHEAD, SEQ);
        tdm_load_tile_2d(lds_addr_of(&vtile[0][0]), V, 64, 32, DHEAD, SEQ, DHEAD);
    }
#endif
    for (int it = 0; it < SEQ / 32; ++it) {
        const int k0 = it * 32;
        const int cur = it & 1;
#if HAVE_TDM
        if (wave == 0) __builtin_amdgcn_s_wait_tensorcnt((short)0);
        __syncthreads();
        if (wave == 0 && it + 1 < SEQ / 32) {
            tdm_load_tile_2d(lds_addr_of(&ktile[1 - cur][0]),
                             Kt + (k0 + 32), 32, 64, SEQ, DHEAD, SEQ);
            tdm_load_tile_2d(lds_addr_of(&vtile[1 - cur][0]),
                             V + (size_t)(k0 + 32) * DHEAD, 64, 32, DHEAD, SEQ, DHEAD);
        }
#else
        // Fallback: cooperative staging of both 4 KB tiles by 256 threads.
        {
            const int t = threadIdx.x;  // 0..255
            // Kt tile [64][32]: thread t covers row t/4, cols (t&3)*8..+7
            *(bf16x8*)&ktile[cur][(t >> 2) * 32 + (t & 3) * 8] =
                *(const bf16x8*)(Kt + (size_t)(t >> 2) * SEQ + k0 + (t & 3) * 8);
            // V tile [32][64]: thread t covers row t/8, cols (t&7)*8..+7
            *(bf16x8*)&vtile[cur][(t >> 3) * 64 + (t & 7) * 8] =
                *(const bf16x8*)(V + (size_t)(k0 + (t >> 3)) * DHEAD + (t & 7) * 8);
            __syncthreads();
        }
#endif
        const __bf16* kt_l = &ktile[cur][0];
#pragma unroll
        for (int t = 0; t < 2; ++t) {
            // Kt B-tiles from LDS: lane = key column lane%16; elem s -> d in table order.
            bf16x16 b0, b1;
#pragma unroll
            for (int s = 0; s < 16; ++s) {
                const int kk = (s < 8) ? (8 * ah + s) : (16 + 8 * ah + s - 8);
                b0[s] = kt_l[kk * 32 + 16 * t + (lane & 15)];
                b1[s] = kt_l[(32 + kk) * 32 + 16 * t + (lane & 15)];
            }
            f32x8 c = {};
            c = wmma_bf16(a0, b0, c);
            c = wmma_bf16(a1, b1, c);
#pragma unroll
            for (int r = 0; r < 8; ++r) {
                // row max over this 16-key tile (reduce across 16-lane group)
                float mt = c[r];
                mt = fmaxf(mt, __shfl_xor(mt, 1, 32));
                mt = fmaxf(mt, __shfl_xor(mt, 2, 32));
                mt = fmaxf(mt, __shfl_xor(mt, 4, 32));
                mt = fmaxf(mt, __shfl_xor(mt, 8, 32));
                const float mnew  = fmaxf(mrow[r], mt);
                const float alpha = __expf(mrow[r] - mnew);
                const float p     = __expf(c[r] - mnew);
                float s = p;
                s += __shfl_xor(s, 1, 32);
                s += __shfl_xor(s, 2, 32);
                s += __shfl_xor(s, 4, 32);
                s += __shfl_xor(s, 8, 32);
                lsum[r] = lsum[r] * alpha + s;
                mrow[r] = mnew;
                o0[r] *= alpha; o1[r] *= alpha; o2[r] *= alpha; o3[r] *= alpha;
                // C/D layout -> row-major [16][32] bf16 in LDS (own wave slice)
                lw[(r + 8 * ah) * 32 + 16 * t + (lane & 15)] = (__bf16)p;
            }
        }
        // CDNA5 split counter: drain LDS stores before re-reading in A layout.
        asm volatile("s_wait_dscnt 0" ::: "memory");
        bf16x16 pa = load_a_pair(lw + am * 32 + 8 * ah, lw + am * 32 + 16 + 8 * ah);

        // V B-tiles from LDS: lane = d column lane%16; elem s -> key in table order.
        const __bf16* v_l = &vtile[cur][lane & 15];
        bf16x16 v0, v1, v2, v3;
#pragma unroll
        for (int s = 0; s < 16; ++s) {
            const int kk = (s < 8) ? (8 * ah + s) : (16 + 8 * ah + s - 8);
            v0[s] = v_l[kk * 64];
            v1[s] = v_l[kk * 64 + 16];
            v2[s] = v_l[kk * 64 + 32];
            v3[s] = v_l[kk * 64 + 48];
        }
        o0 = wmma_bf16(pa, v0, o0);
        o1 = wmma_bf16(pa, v1, o1);
        o2 = wmma_bf16(pa, v2, o2);
        o3 = wmma_bf16(pa, v3, o3);
        __syncthreads();   // all waves done with tile[cur] before it+2 overwrite
    }

    // normalize + store ctx[b*SEQ+q, h*64+d] as bf16
    const int b = bh >> 4;
    const int h = bh & (NHEAD - 1);
    f32x8 os[4] = {o0, o1, o2, o3};
#pragma unroll
    for (int r = 0; r < 8; ++r) {
        const float inv = 1.0f / lsum[r];
        const int m = q0 + r + 8 * (lane >> 4);
        const size_t rowbase = (size_t)(b * SEQ + m) * D_MODEL + h * DHEAD;
#pragma unroll
        for (int j = 0; j < 4; ++j) {
            ctx16[rowbase + 16 * j + (lane & 15)] = (__bf16)(os[j][r] * inv);
        }
    }
}

// ---------------------------------------------------------------------------
// Host-side orchestration.
// ---------------------------------------------------------------------------
extern "C" void kernel_launch(void* const* d_in, const int* in_sizes, int n_in,
                              void* d_out, int out_size, void* d_ws, size_t ws_size,
                              hipStream_t stream) {
    const float* dec = (const float*)d_in[0];
    const float* enc = (const float*)d_in[1];
    const float* wq  = (const float*)d_in[2];
    const float* bq  = (const float*)d_in[3];
    const float* wk  = (const float*)d_in[4];
    const float* bk  = (const float*)d_in[5];
    const float* wv  = (const float*)d_in[6];
    const float* bv  = (const float*)d_in[7];
    const float* wo  = (const float*)d_in[8];
    const float* bo  = (const float*)d_in[9];
    float* outf = (float*)d_out;

    const size_t ACT  = (size_t)MROWS * D_MODEL;      // 4194304 elements
    const size_t WELT = (size_t)D_MODEL * D_MODEL;    // 1048576 elements

    char* ws = (char*)d_ws;
    size_t off = 0;
    auto take = [&](size_t elems) { __bf16* p = (__bf16*)(ws + off); off += elems * 2; return p; };
    __bf16* dec16 = take(ACT);
    __bf16* enc16 = take(ACT);
    __bf16* wq16  = take(WELT);
    __bf16* wk16  = take(WELT);
    __bf16* wv16  = take(WELT);
    __bf16* wo16  = take(WELT);
    __bf16* q16   = take(ACT);   // [B,H,SQ,64]
    __bf16* kt16  = take(ACT);   // [B,H,64,SK]
    __bf16* v16   = take(ACT);   // [B,H,SK,64]
    __bf16* ctx16 = take(ACT);   // [4096,1024]
    (void)ws_size; (void)in_sizes; (void)n_in; (void)out_size;

    // 1) fp32 -> bf16
    cvt_f32_bf16<<<dim3((unsigned)(ACT / 1024)), 256, 0, stream>>>(dec, dec16, (int)(ACT / 4));
    cvt_f32_bf16<<<dim3((unsigned)(ACT / 1024)), 256, 0, stream>>>(enc, enc16, (int)(ACT / 4));
    cvt_f32_bf16<<<dim3((unsigned)(WELT / 1024)), 256, 0, stream>>>(wq, wq16, (int)(WELT / 4));
    cvt_f32_bf16<<<dim3((unsigned)(WELT / 1024)), 256, 0, stream>>>(wk, wk16, (int)(WELT / 4));
    cvt_f32_bf16<<<dim3((unsigned)(WELT / 1024)), 256, 0, stream>>>(wv, wv16, (int)(WELT / 4));
    cvt_f32_bf16<<<dim3((unsigned)(WELT / 1024)), 256, 0, stream>>>(wo, wo16, (int)(WELT / 4));

    const dim3 ggrid(D_MODEL / 64, MROWS / 128);  // (16, 32)
    // 2) projections
    gemm_bf16<<<ggrid, 256, 0, stream>>>(dec16, wq16, bq, (void*)q16, 0);
    gemm_bf16<<<ggrid, 256, 0, stream>>>(enc16, wk16, bk, (void*)kt16, 1);
    gemm_bf16<<<ggrid, 256, 0, stream>>>(enc16, wv16, bv, (void*)v16, 2);
    // 3) attention
    flash_attn<<<dim3(SEQ / 128, BATCH * NHEAD), 256, 0, stream>>>(q16, kt16, v16, ctx16);
    // 4) output projection (fp32 out + bias)
    gemm_bf16<<<ggrid, 256, 0, stream>>>(ctx16, wo16, bo, (void*)outf, 3);
}
